// Agent_8598524527072
// MI455X (gfx1250) — compile-verified, matrix operand order checked by reading, not run on hardware
//
#include <hip/hip_runtime.h>
#include <hip/hip_bf16.h>

// ---------------------------------------------------------------------------
// Problem constants (from the reference)
// ---------------------------------------------------------------------------
constexpr int Lz    = 4;
constexpr int Bz    = 32;
constexpr int Tz    = 128;
constexpr int MEMz  = 128;
constexpr int Dz    = 1024;
constexpr int Hz    = 16;
constexpr int HDz   = 64;
constexpr int KLz   = MEMz + Tz;     // 256
constexpr int MAXDz = KLz + Tz;      // 384
constexpr int Nflat = Bz * Tz * Dz;  // 4,194,304

typedef __bf16 bf16;
typedef __attribute__((ext_vector_type(16))) __bf16 v16bf;
typedef __attribute__((ext_vector_type(8)))  __bf16 v8bf;
typedef __attribute__((ext_vector_type(8)))  float  v8f;

// ---------------------------------------------------------------------------
// WMMA helpers (wave32, 16x16x32 bf16 -> f32)
// ---------------------------------------------------------------------------
__device__ __forceinline__ v8f vzero8() {
  v8f z;
#pragma unroll
  for (int i = 0; i < 8; ++i) z[i] = 0.0f;
  return z;
}

__device__ __forceinline__ v8f wmma_bf16(v16bf a, v16bf b, v8f c) {
  return __builtin_amdgcn_wmma_f32_16x16x32_bf16(
      /*neg_a=*/false, a, /*neg_b=*/false, b,
      /*c_mod=*/(short)0, c, /*reuse_a=*/false, /*reuse_b=*/false);
}

// Load a 16x32 bf16 fragment from a row-major matrix.
// ISA 16-bit A layout: lanes 0-15 hold K=0..7 & 16..23, lanes 16-31 hold
// K=8..15 & 24..31 of row (lane%16). Two contiguous 16B chunks per lane.
// The B operand (for C = A @ W^T) is loaded with the identical pattern from
// rows of W.
__device__ __forceinline__ v16bf frag_ld(const bf16* __restrict__ row0,
                                         int ld, int kb) {
  const int lane = threadIdx.x & 31;
  const bf16* p = row0 + (size_t)(lane & 15) * ld + kb + ((lane >> 4) << 3);
  v8bf lo = *(const v8bf*)(p);
  v8bf hi = *(const v8bf*)(p + 16);
  v16bf r;
#pragma unroll
  for (int i = 0; i < 8; ++i) { r[i] = lo[i]; r[i + 8] = hi[i]; }
  return r;
}

// CDNA5 async global->LDS copy (ASYNCcnt-tracked).  The generic address of a
// __shared__ object carries the LDS byte offset in its low 32 bits
__device__ __forceinline__ void async_ld_b128(void* lds, const void* gptr) {
  unsigned loff = (unsigned)(uintptr_t)lds;
  asm volatile("global_load_async_to_lds_b128 %0, %1, off"
               :: "v"(loff), "v"(gptr) : "memory");
}
__device__ __forceinline__ void wait_async0() {
  asm volatile("s_wait_asynccnt 0" ::: "memory");
}

// C/D fragment layout: lane l, vgpr g -> m = g + 8*(l/16), n = l%16.

// ---------------------------------------------------------------------------
// LayerNorm (one wave per 1024-wide row), optional concat source (mem | h)
// ---------------------------------------------------------------------------
__global__ __launch_bounds__(256) void ln_kernel(
    const float* __restrict__ srcA, const float* __restrict__ srcB,
    int rowsA, int rowsPer, int rowsBstride,
    const float* __restrict__ gam, const float* __restrict__ bet,
    bf16* __restrict__ out) {
  const int wv = threadIdx.x >> 5, lane = threadIdx.x & 31;
  const int row = blockIdx.x * 8 + wv;
  const int b = row / rowsPer, k = row % rowsPer;
  const float* src = (k < rowsA)
      ? (srcA + ((size_t)b * rowsA + k) * Dz)
      : (srcB + ((size_t)b * rowsBstride + (k - rowsA)) * Dz);
  float v[32];
  float s = 0.f;
#pragma unroll
  for (int i = 0; i < 32; ++i) { v[i] = src[lane + i * 32]; s += v[i]; }
#pragma unroll
  for (int off = 16; off; off >>= 1) s += __shfl_xor(s, off, 32);
  const float mu = s * (1.0f / 1024.0f);
  float q = 0.f;
#pragma unroll
  for (int i = 0; i < 32; ++i) { float d = v[i] - mu; q += d * d; }
#pragma unroll
  for (int off = 16; off; off >>= 1) q += __shfl_xor(q, off, 32);
  const float rstd = rsqrtf(q * (1.0f / 1024.0f) + 1e-5f);
  const size_t ro = (size_t)row * Dz;
#pragma unroll
  for (int i = 0; i < 32; ++i) {
    const int j = lane + i * 32;
    out[ro + j] = (bf16)((v[i] - mu) * rstd * gam[j] + bet[j]);
  }
}

// ---------------------------------------------------------------------------
// Q head projection: qh = qn_head @ Wq^T ; qhu = qh+u, qhv = qh+v (bf16)
// layouts: qn [B,T,D]; out [B,H,T,HD]
// ---------------------------------------------------------------------------
__global__ __launch_bounds__(256) void proj_q_kernel(
    const bf16* __restrict__ qn, const bf16* __restrict__ Wq,
    const float* __restrict__ u, const float* __restrict__ v,
    bf16* __restrict__ qhu, bf16* __restrict__ qhv) {
  const int gw = blockIdx.x * 8 + (threadIdx.x >> 5);
  const int qt = gw & 7, h = (gw >> 3) & 15, b = gw >> 7;
  const bf16* A = qn + ((size_t)(b * Tz + qt * 16)) * Dz + h * HDz;
  v8f acc[4];
#pragma unroll
  for (int j = 0; j < 4; ++j) acc[j] = vzero8();
  for (int ks = 0; ks < 2; ++ks) {
    v16bf a = frag_ld(A, Dz, ks * 32);
#pragma unroll
    for (int j = 0; j < 4; ++j)
      acc[j] = wmma_bf16(a, frag_ld(Wq + j * 16 * HDz, HDz, ks * 32), acc[j]);
  }
  const int lane = threadIdx.x & 31;
  const int n0 = lane & 15, mh = (lane >> 4) * 8;
#pragma unroll
  for (int j = 0; j < 4; ++j) {
#pragma unroll
    for (int g = 0; g < 8; ++g) {
      const int n = j * 16 + n0;
      const size_t o = ((size_t)(b * Hz + h) * Tz + qt * 16 + mh + g) * HDz + n;
      const float t = acc[j][g];
      qhu[o] = (bf16)(t + u[h * HDz + n]);
      qhv[o] = (bf16)(t + v[h * HDz + n]);
    }
  }
}

// ---------------------------------------------------------------------------
// K/V head projection: kh [B,H,K,HD], vhT [B,H,HD,K] (V stored transposed)
// ---------------------------------------------------------------------------
__global__ __launch_bounds__(256) void proj_kv_kernel(
    const bf16* __restrict__ kvn, const bf16* __restrict__ Wk,
    const bf16* __restrict__ Wv, bf16* __restrict__ kh,
    bf16* __restrict__ vhT) {
  const int gw = blockIdx.x * 8 + (threadIdx.x >> 5);
  const int kt = gw & 15, h = (gw >> 4) & 15, b = gw >> 8;
  const bf16* A = kvn + ((size_t)(b * KLz + kt * 16)) * Dz + h * HDz;
  v8f ak[4], av[4];
#pragma unroll
  for (int j = 0; j < 4; ++j) { ak[j] = vzero8(); av[j] = vzero8(); }
  for (int ks = 0; ks < 2; ++ks) {
    v16bf a = frag_ld(A, Dz, ks * 32);
#pragma unroll
    for (int j = 0; j < 4; ++j) {
      ak[j] = wmma_bf16(a, frag_ld(Wk + j * 16 * HDz, HDz, ks * 32), ak[j]);
      av[j] = wmma_bf16(a, frag_ld(Wv + j * 16 * HDz, HDz, ks * 32), av[j]);
    }
  }
  const int lane = threadIdx.x & 31;
  const int n0 = lane & 15, mh = (lane >> 4) * 8;
#pragma unroll
  for (int j = 0; j < 4; ++j) {
#pragma unroll
    for (int g = 0; g < 8; ++g) {
      const int n = j * 16 + n0;
      const int krow = kt * 16 + mh + g;
      kh[((size_t)(b * Hz + h) * KLz + krow) * HDz + n] = (bf16)ak[j][g];
      vhT[((size_t)(b * Hz + h) * HDz + n) * KLz + krow] = (bf16)av[j][g];
    }
  }
}

// ---------------------------------------------------------------------------
// rel = (pos_emb @ Wr^T) rearranged to [H, MAXD, HD] bf16
// ---------------------------------------------------------------------------
__global__ __launch_bounds__(256) void gemm_rel_kernel(
    const bf16* __restrict__ pos, const bf16* __restrict__ Wr,
    bf16* __restrict__ rel) {
  const int gw = blockIdx.x * 8 + (threadIdx.x >> 5);  // 384 waves
  const int mt = gw % 24, nb = gw / 24;                // nb == head
  const bf16* A = pos + (size_t)mt * 16 * Dz;
  v8f acc[4];
#pragma unroll
  for (int j = 0; j < 4; ++j) acc[j] = vzero8();
  for (int ks = 0; ks < 32; ++ks) {
    v16bf a = frag_ld(A, Dz, ks * 32);
#pragma unroll
    for (int j = 0; j < 4; ++j)
      acc[j] = wmma_bf16(
          a, frag_ld(Wr + (size_t)(nb * 64 + j * 16) * Dz, Dz, ks * 32),
          acc[j]);
  }
  const int lane = threadIdx.x & 31;
  const int n0 = lane & 15, mh = (lane >> 4) * 8;
#pragma unroll
  for (int j = 0; j < 4; ++j)
#pragma unroll
    for (int g = 0; g < 8; ++g)
      rel[((size_t)nb * MAXDz + mt * 16 + mh + g) * HDz + j * 16 + n0] =
          (bf16)acc[j][g];
}

// ---------------------------------------------------------------------------
// Fused attention per (b,h): E = (q+u)K^T + shift-gather((q+v)R^T), causal
// mask, softmax, P@V.  8 waves/WG; 160KB LDS:
//   BD/P region (96KB) + async-staged K (32KB) + V^T (32KB).
// ---------------------------------------------------------------------------
__global__ __launch_bounds__(256) void attn_kernel(
    const bf16* __restrict__ qhu, const bf16* __restrict__ qhv,
    const bf16* __restrict__ kh, const bf16* __restrict__ vhT,
    const bf16* __restrict__ rel, bf16* __restrict__ attout) {
  __shared__ bf16 smem[Tz * MAXDz];   // 96 KB: BD, later P staging
  __shared__ bf16 kv_s[KLz * HDz];    // 32 KB: K head tile
  __shared__ bf16 vT_s[HDz * KLz];    // 32 KB: V^T head tile
  const int bh = blockIdx.x;
  const int b = bh >> 4, h = bh & 15;
  const int wv = threadIdx.x >> 5, lane = threadIdx.x & 31;
  const int n0 = lane & 15, mh = (lane >> 4) * 8;

  const bf16* qhu_b = qhu + (size_t)bh * Tz * HDz;
  const bf16* qhv_b = qhv + (size_t)bh * Tz * HDz;
  const bf16* kh_b  = kh  + (size_t)bh * KLz * HDz;
  const bf16* vh_b  = vhT + (size_t)bh * HDz * KLz;
  const bf16* rel_h = rel + (size_t)h * MAXDz * HDz;

  // Kick off async staging of K / V^T into LDS; overlaps with BD WMMAs.
  {
    const int tid = threadIdx.x;
#pragma unroll
    for (int i = 0; i < 8; ++i) {
      const int c = (tid + i * 256) * 16;  // byte offset of 16B chunk
      async_ld_b128((char*)kv_s + c, (const char*)kh_b + c);
      async_ld_b128((char*)vT_s + c, (const char*)vh_b + c);
    }
  }

  // Phase A: BD[q][r] = (q+v) . rel[r]   (8 x 24 tiles of 16x16)
  for (int t = wv; t < 192; t += 8) {
    const int qt = t & 7, rt = t >> 3;
    v8f acc = vzero8();
    for (int ks = 0; ks < 2; ++ks)
      acc = wmma_bf16(frag_ld(qhv_b + qt * 16 * HDz, HDz, ks * 32),
                      frag_ld(rel_h + rt * 16 * HDz, HDz, ks * 32), acc);
#pragma unroll
    for (int g = 0; g < 8; ++g)
      smem[(size_t)(qt * 16 + mh + g) * MAXDz + rt * 16 + n0] = (bf16)acc[g];
  }
  wait_async0();
  __syncthreads();  // BD visible + K/V^T staged

  // Phase B: each wave owns q-strip [wv*16, wv*16+16) x K=256 energies
  v8f e[16];
#pragma unroll
  for (int j = 0; j < 16; ++j) e[j] = vzero8();
  {
    const v16bf a0 = frag_ld(qhu_b + wv * 16 * HDz, HDz, 0);
    const v16bf a1 = frag_ld(qhu_b + wv * 16 * HDz, HDz, 32);
#pragma unroll
    for (int j = 0; j < 16; ++j) {
      e[j] = wmma_bf16(a0, frag_ld(kv_s + j * 16 * HDz, HDz, 0), e[j]);
      e[j] = wmma_bf16(a1, frag_ld(kv_s + j * 16 * HDz, HDz, 32), e[j]);
    }
  }
  float rowmax[8], rowsum[8];
#pragma unroll
  for (int g = 0; g < 8; ++g) rowmax[g] = -3.0e38f;
#pragma unroll
  for (int j = 0; j < 16; ++j) {
#pragma unroll
    for (int g = 0; g < 8; ++g) {
      const int q = wv * 16 + mh + g;
      const int k = j * 16 + n0;
      const int idx = q + MEMz - k;  // rel distance; >=0 <=> unmasked
      float val;
      if (idx < 0) val = -1.0e30f;
      else val = e[j][g] + (float)smem[(size_t)q * MAXDz + idx];
      e[j][g] = val;
      rowmax[g] = fmaxf(rowmax[g], val);
    }
  }
#pragma unroll
  for (int g = 0; g < 8; ++g)
#pragma unroll
    for (int off = 1; off < 16; off <<= 1)
      rowmax[g] = fmaxf(rowmax[g], __shfl_xor(rowmax[g], off, 32));
#pragma unroll
  for (int g = 0; g < 8; ++g) rowsum[g] = 0.f;
  const float scale = 0.125f;  // 1/sqrt(64)
#pragma unroll
  for (int j = 0; j < 16; ++j)
#pragma unroll
    for (int g = 0; g < 8; ++g) {
      const float p = __expf((e[j][g] - rowmax[g]) * scale);
      e[j][g] = p;
      rowsum[g] += p;
    }
#pragma unroll
  for (int g = 0; g < 8; ++g) {
#pragma unroll
    for (int off = 1; off < 16; off <<= 1)
      rowsum[g] += __shfl_xor(rowsum[g], off, 32);
    rowsum[g] = 1.0f / rowsum[g];
  }
  __syncthreads();  // all waves done with BD; reuse LDS for P staging

  bf16* P = smem + wv * 16 * 256;  // 8 waves * 8KB = 64KB <= 96KB
#pragma unroll
  for (int j = 0; j < 16; ++j)
#pragma unroll
    for (int g = 0; g < 8; ++g)
      P[(mh + g) * 256 + j * 16 + n0] = (bf16)(e[j][g] * rowsum[g]);
  __syncthreads();

  // Phase C: out strip (16 x 64) = P (16 x 256) @ V (256 x 64)
  v8f o[4];
#pragma unroll
  for (int j = 0; j < 4; ++j) o[j] = vzero8();
  for (int ks = 0; ks < 8; ++ks) {
    v16bf a = frag_ld(P, 256, ks * 32);
#pragma unroll
    for (int j = 0; j < 4; ++j)
      o[j] = wmma_bf16(a, frag_ld(vT_s + (size_t)j * 16 * KLz, KLz, ks * 32),
                       o[j]);
  }
#pragma unroll
  for (int j = 0; j < 4; ++j)
#pragma unroll
    for (int g = 0; g < 8; ++g)
      attout[((size_t)(b * Tz + wv * 16 + mh + g)) * Dz + h * HDz + j * 16 +
             n0] = (bf16)o[j][g];
}

// ---------------------------------------------------------------------------
// General (optionally dual) GEMM: C = act(A1@B1^T [+ A2@B2^T] + bsign*bias)
// M=4096, N=1024, Kd=1024. ACT: 0=id, 1=relu, 2=sigmoid, 3=tanh.
// Each wave owns a 64(M) x 64(N) tile: 16 accumulators, 8 fragment loads per
// k-step feed 16 WMMAs (0.5 loads/WMMA). All config is compile-time so the
// 128-element epilogue is branch-free straight-line code.
// ---------------------------------------------------------------------------
template <bool DUAL, int ACT, bool HASBIAS, bool OUTF, bool OUTB>
__global__ __launch_bounds__(256) void gemm_t_kernel(
    const bf16* __restrict__ A1, const bf16* __restrict__ B1,
    const bf16* __restrict__ A2, const bf16* __restrict__ B2,
    const float* __restrict__ bias, float bsign,
    float* __restrict__ outF, bf16* __restrict__ outB) {
  const int gw = blockIdx.x * 8 + (threadIdx.x >> 5);  // 1024 waves
  const int mt = gw & 63, nb = gw >> 6;
  v8f acc[4][4];
#pragma unroll
  for (int i = 0; i < 4; ++i)
#pragma unroll
    for (int j = 0; j < 4; ++j) acc[i][j] = vzero8();

#pragma unroll
  for (int s = 0; s < (DUAL ? 2 : 1); ++s) {
    const bf16* Ab = (s == 0 ? A1 : A2) + (size_t)mt * 64 * Dz;
    const bf16* Bb = (s == 0 ? B1 : B2) + (size_t)nb * 64 * Dz;
    for (int ks = 0; ks < 32; ++ks) {
      if (ks + 1 < 32)  // gfx1250 speculative prefetch of next weight chunk
        __builtin_prefetch(Bb + (ks + 1) * 32, 0, 0);
      v16bf a[4], bw[4];
#pragma unroll
      for (int i = 0; i < 4; ++i)
        a[i] = frag_ld(Ab + (size_t)i * 16 * Dz, Dz, ks * 32);
#pragma unroll
      for (int j = 0; j < 4; ++j)
        bw[j] = frag_ld(Bb + (size_t)j * 16 * Dz, Dz, ks * 32);
#pragma unroll
      for (int i = 0; i < 4; ++i)
#pragma unroll
        for (int j = 0; j < 4; ++j)
          acc[i][j] = wmma_bf16(a[i], bw[j], acc[i][j]);
    }
  }
  const int lane = threadIdx.x & 31;
  const int n0 = lane & 15, mh = (lane >> 4) * 8;
  float bvals[4];
#pragma unroll
  for (int j = 0; j < 4; ++j)
    bvals[j] = HASBIAS ? bsign * bias[nb * 64 + j * 16 + n0] : 0.0f;
#pragma unroll
  for (int i = 0; i < 4; ++i) {
#pragma unroll
    for (int g = 0; g < 8; ++g) {
      const size_t row =
          (size_t)(mt * 64 + i * 16 + mh + g) * Dz + nb * 64 + n0;
#pragma unroll
      for (int j = 0; j < 4; ++j) {
        float t = acc[i][j][g];
        if (HASBIAS) t += bvals[j];
        if (ACT == 1) t = fmaxf(t, 0.0f);
        else if (ACT == 2) t = 1.0f / (1.0f + __expf(-t));
        else if (ACT == 3) t = tanhf(t);
        if (OUTF) outF[row + j * 16] = t;
        if (OUTB) outB[row + j * 16] = (bf16)t;
      }
    }
  }
}

// ---------------------------------------------------------------------------
// Elementwise kernels
// ---------------------------------------------------------------------------
__global__ __launch_bounds__(256) void posemb_kernel(bf16* __restrict__ pos) {
  const int i = blockIdx.x * 256 + threadIdx.x;
  if (i >= MAXDz * Dz) return;
  const int r = i / Dz, c = i % Dz;
  const int c2 = (c < 512) ? c : c - 512;
  const float f = __powf(10000.0f, -(float)(2 * c2) / (float)Dz);
  const float a = (float)r * f;
  pos[i] = (bf16)((c < 512) ? __sinf(a) : __cosf(a));
}

__global__ __launch_bounds__(256) void copy_xh_kernel(
    const float* __restrict__ x, float* __restrict__ hf,
    bf16* __restrict__ hb, int n) {
  const int i = blockIdx.x * 256 + threadIdx.x;
  if (i >= n) return;
  const float v = x[i];
  hf[i] = v;
  hb[i] = (bf16)v;
}

__global__ __launch_bounds__(256) void ew_rx_kernel(
    const float* __restrict__ r, const float* __restrict__ x,
    bf16* __restrict__ rx, int n) {
  const int i = blockIdx.x * 256 + threadIdx.x;
  if (i >= n) return;
  rx[i] = (bf16)(r[i] * x[i]);
}

__global__ __launch_bounds__(256) void ew_combine_kernel(
    const float* __restrict__ z, const float* __restrict__ x,
    const float* __restrict__ hc, float* __restrict__ of,
    bf16* __restrict__ ob, int n) {
  const int i = blockIdx.x * 256 + threadIdx.x;
  if (i >= n) return;
  const float zi = z[i];
  const float o = (1.0f - zi) * x[i] + zi * hc[i];
  of[i] = o;
  ob[i] = (bf16)o;
}

__global__ __launch_bounds__(256) void cvt_kernel(const float* __restrict__ s,
                                                  bf16* __restrict__ d,
                                                  int n) {
  const int i = blockIdx.x * 256 + threadIdx.x;
  if (i >= n) return;
  d[i] = (bf16)s[i];
}

// ---------------------------------------------------------------------------
// Host launcher
// ---------------------------------------------------------------------------
extern "C" void kernel_launch(void* const* d_in, const int* in_sizes, int n_in,
                              void* d_out, int out_size, void* d_ws,
                              size_t ws_size, hipStream_t stream) {
  (void)in_sizes; (void)n_in; (void)ws_size;
  // Inputs, in setup_inputs() dict order (params dict in insertion order).
  const float* x        = (const float*)d_in[0];
  const float* memories = (const float*)d_in[1];
  // d_in[2] = mask (deterministic causal mask; computed in-kernel instead)
  const float* ln_q_s = (const float*)d_in[3];
  const float* ln_q_b = (const float*)d_in[4];
  const float* ln_kv_s = (const float*)d_in[5];
  const float* ln_kv_b = (const float*)d_in[6];
  const float* ln_f_s = (const float*)d_in[7];
  const float* ln_f_b = (const float*)d_in[8];
  const float* Wv = (const float*)d_in[9];
  const float* Wk = (const float*)d_in[10];
  const float* Wq = (const float*)d_in[11];
  const float* Wr = (const float*)d_in[12];
  const float* uP = (const float*)d_in[13];
  const float* vP = (const float*)d_in[14];
  const float* Wo = (const float*)d_in[15];
  const float* bo = (const float*)d_in[16];
  const float* Wfc = (const float*)d_in[17];
  const float* bfc = (const float*)d_in[18];
  const float* ag_wr = (const float*)d_in[19];
  const float* ag_ur = (const float*)d_in[20];
  const float* ag_wz = (const float*)d_in[21];
  const float* ag_uz = (const float*)d_in[22];
  const float* ag_wg = (const float*)d_in[23];
  const float* ag_ug = (const float*)d_in[24];
  const float* ag_bg = (const float*)d_in[25];
  const float* fg_wr = (const float*)d_in[26];
  const float* fg_ur = (const float*)d_in[27];
  const float* fg_wz = (const float*)d_in[28];
  const float* fg_uz = (const float*)d_in[29];
  const float* fg_wg = (const float*)d_in[30];
  const float* fg_ug = (const float*)d_in[31];
  const float* fg_bg = (const float*)d_in[32];

  char* p = (char*)d_ws;
  auto alloc = [&](size_t bytes) -> void* {
    void* r = (void*)p;
    p += (bytes + 255) & ~(size_t)255;
    return r;
  };
  float* h_f32 = (float*)alloc((size_t)Nflat * 4);
  bf16* h_b = (bf16*)alloc((size_t)Nflat * 2);
  bf16* qn = (bf16*)alloc((size_t)Nflat * 2);
  bf16* kvn = (bf16*)alloc((size_t)Bz * KLz * Dz * 2);
  bf16* qhu = (bf16*)alloc((size_t)Bz * Hz * Tz * HDz * 2);
  bf16* qhv = (bf16*)alloc((size_t)Bz * Hz * Tz * HDz * 2);
  bf16* khb = (bf16*)alloc((size_t)Bz * Hz * KLz * HDz * 2);
  bf16* vhT = (bf16*)alloc((size_t)Bz * Hz * HDz * KLz * 2);
  bf16* relb = (bf16*)alloc((size_t)Hz * MAXDz * HDz * 2);
  bf16* posb = (bf16*)alloc((size_t)MAXDz * Dz * 2);
  bf16* attb = (bf16*)alloc((size_t)Nflat * 2);
  bf16* yb = (bf16*)alloc((size_t)Nflat * 2);
  float* rf = (float*)alloc((size_t)Nflat * 4);
  float* zf = (float*)alloc((size_t)Nflat * 4);
  float* hcf = (float*)alloc((size_t)Nflat * 4);
  bf16* rxb = (bf16*)alloc((size_t)Nflat * 2);
  float* xgf = (float*)alloc((size_t)Nflat * 4);
  bf16* xgb = (bf16*)alloc((size_t)Nflat * 2);
  bf16* flnb = (bf16*)alloc((size_t)Nflat * 2);
  bf16* y2b = (bf16*)alloc((size_t)Nflat * 2);
  bf16* wbig[6];
  for (int i = 0; i < 6; ++i) wbig[i] = (bf16*)alloc((size_t)Dz * Dz * 2);
  bf16* wsm[3];
  for (int i = 0; i < 3; ++i) wsm[i] = (bf16*)alloc((size_t)HDz * HDz * 2);

  const int ewBlocks = (Nflat + 255) / 256;

  posemb_kernel<<<(MAXDz * Dz + 255) / 256, 256, 0, stream>>>(posb);
  copy_xh_kernel<<<ewBlocks, 256, 0, stream>>>(x, h_f32, h_b, Nflat);

  const size_t DD = (size_t)Dz * Dz;
  const size_t HH = (size_t)HDz * HDz;

  auto cvt = [&](const float* src, bf16* dst, size_t n) {
    cvt_kernel<<<((int)n + 255) / 256, 256, 0, stream>>>(src, dst, (int)n);
  };

  for (int l = 0; l < Lz; ++l) {
    // ---- stage this layer's weights as bf16 ----
    cvt(Wq + l * HH, wsm[0], HH);
    cvt(Wk + l * HH, wsm[1], HH);
    cvt(Wv + l * HH, wsm[2], HH);
    cvt(Wr + l * DD, wbig[0], DD);

    // ---- LayerNorms ----
    ln_kernel<<<(Bz * Tz) / 8, 256, 0, stream>>>(
        h_f32, nullptr, Tz, Tz, Tz, ln_q_s + l * Dz, ln_q_b + l * Dz, qn);
    ln_kernel<<<(Bz * KLz) / 8, 256, 0, stream>>>(
        memories + (size_t)l * Bz * MEMz * Dz, h_f32, MEMz, KLz, Tz,
        ln_kv_s + l * Dz, ln_kv_b + l * Dz, kvn);

    // ---- head projections + rel GEMM ----
    proj_q_kernel<<<(Bz * Hz * 8) / 8, 256, 0, stream>>>(
        qn, wsm[0], uP + (size_t)l * Hz * HDz, vP + (size_t)l * Hz * HDz, qhu,
        qhv);
    proj_kv_kernel<<<(Bz * Hz * 16) / 8, 256, 0, stream>>>(kvn, wsm[1], wsm[2],
                                                           khb, vhT);
    gemm_rel_kernel<<<(24 * 16) / 8, 256, 0, stream>>>(posb, wbig[0], relb);

    // ---- fused attention ----
    attn_kernel<<<Bz * Hz, 256, 0, stream>>>(qhu, qhv, khb, vhT, relb, attb);

    // ---- output proj + relu (y = relu(att@Wo^T + bo), bf16) ----
    cvt(Wo + l * DD, wbig[0], DD);
    gemm_t_kernel<false, 1, true, false, true><<<128, 256, 0, stream>>>(
        attb, wbig[0], nullptr, nullptr, bo + l * Dz, 1.0f, nullptr, yb);

    // ---- attention GRU gate: xg = gru(h, y) ----
    cvt(ag_wr + l * DD, wbig[0], DD);
    cvt(ag_ur + l * DD, wbig[1], DD);
    cvt(ag_wz + l * DD, wbig[2], DD);
    cvt(ag_uz + l * DD, wbig[3], DD);
    cvt(ag_wg + l * DD, wbig[4], DD);
    cvt(ag_ug + l * DD, wbig[5], DD);
    gemm_t_kernel<true, 2, false, true, false><<<128, 256, 0, stream>>>(
        yb, wbig[0], h_b, wbig[1], nullptr, 0.0f, rf, nullptr);
    gemm_t_kernel<true, 2, true, true, false><<<128, 256, 0, stream>>>(
        yb, wbig[2], h_b, wbig[3], ag_bg + l * Dz, -1.0f, zf, nullptr);
    ew_rx_kernel<<<ewBlocks, 256, 0, stream>>>(rf, h_f32, rxb, Nflat);
    gemm_t_kernel<true, 3, false, true, false><<<128, 256, 0, stream>>>(
        yb, wbig[4], rxb, wbig[5], nullptr, 0.0f, hcf, nullptr);
    ew_combine_kernel<<<ewBlocks, 256, 0, stream>>>(zf, h_f32, hcf, xgf, xgb,
                                                    Nflat);

    // ---- FFN: y2 = relu(LN(xg)@Wfc^T + bfc) ----
    ln_kernel<<<(Bz * Tz) / 8, 256, 0, stream>>>(
        xgf, nullptr, Tz, Tz, Tz, ln_f_s + l * Dz, ln_f_b + l * Dz, flnb);
    cvt(Wfc + l * DD, wbig[0], DD);
    gemm_t_kernel<false, 1, true, false, true><<<128, 256, 0, stream>>>(
        flnb, wbig[0], nullptr, nullptr, bfc + l * Dz, 1.0f, nullptr, y2b);

    // ---- FFN GRU gate: h = gru(xg, y2) ----
    cvt(fg_wr + l * DD, wbig[0], DD);
    cvt(fg_ur + l * DD, wbig[1], DD);
    cvt(fg_wz + l * DD, wbig[2], DD);
    cvt(fg_uz + l * DD, wbig[3], DD);
    cvt(fg_wg + l * DD, wbig[4], DD);
    cvt(fg_ug + l * DD, wbig[5], DD);
    gemm_t_kernel<true, 2, false, true, false><<<128, 256, 0, stream>>>(
        y2b, wbig[0], xgb, wbig[1], nullptr, 0.0f, rf, nullptr);
    gemm_t_kernel<true, 2, true, true, false><<<128, 256, 0, stream>>>(
        y2b, wbig[2], xgb, wbig[3], fg_bg + l * Dz, -1.0f, zf, nullptr);
    ew_rx_kernel<<<ewBlocks, 256, 0, stream>>>(rf, xgf, rxb, Nflat);
    gemm_t_kernel<true, 3, false, true, false><<<128, 256, 0, stream>>>(
        y2b, wbig[4], rxb, wbig[5], nullptr, 0.0f, hcf, nullptr);
    ew_combine_kernel<<<ewBlocks, 256, 0, stream>>>(zf, xgf, hcf, h_f32, h_b,
                                                    Nflat);
  }

  hipMemcpyAsync(d_out, h_f32, (size_t)out_size * sizeof(float),
                 hipMemcpyDeviceToDevice, stream);
}
